// EPT_36644660970213
// MI455X (gfx1250) — matrix operand order repaired and checked by reading, not compiled, not run on hardware
//
#include <hip/hip_runtime.h>
#include <cmath>

#define DM 256
#define NHEAD 8
#define NLVL 3
#define NPNT 4
#define BSZ 8
#define STOT 4725
#define NQQ 3600

typedef _Float16 v16h __attribute__((ext_vector_type(16)));
typedef _Float16 v8h  __attribute__((ext_vector_type(8)));
typedef float v8f __attribute__((ext_vector_type(8)));

__device__ __forceinline__ v16h EPT_cat(v8h a, v8h b) {
    return __builtin_shufflevector(a, b, 0,1,2,3,4,5,6,7,8,9,10,11,12,13,14,15);
}

// ---------------------------------------------------------------------------
// Split fp32 -> (hi f16, lo f16) planes. lo = fp16(x - fp32(hi)).
// ---------------------------------------------------------------------------
__global__ void EPT_split16(const float* __restrict__ src, _Float16* __restrict__ hi,
                            _Float16* __restrict__ lo, int n)
{
    int i = blockIdx.x * blockDim.x + threadIdx.x;
    if (i >= n) return;
    float x = src[i];
    _Float16 h = (_Float16)x;
    hi[i] = h;
    lo[i] = (_Float16)(x - (float)h);
}

// ---------------------------------------------------------------------------
// GEMM: Y[M,N] = X[M,K] @ W[N,K]^T + bias[N] (optional ReLU), fp32 result.
// X and W pre-split into f16 hi/lo planes; 3-product compensation gives
// ~fp32 accuracy: hi*hi + hi*lo + lo*hi on v_wmma_f32_16x16x32_f16.
// One wave computes a 32x32 tile (2x2 WMMA tiles): 12 WMMAs per K-step.
// N, K multiples of 32 (true for all layers); M guarded.
// ---------------------------------------------------------------------------
__global__ __launch_bounds__(32) void EPT_gemm_wmma(
    const _Float16* __restrict__ Xh, const _Float16* __restrict__ Xl,
    const _Float16* __restrict__ Wh, const _Float16* __restrict__ Wl,
    const float* __restrict__ bias, float* __restrict__ Y,
    int M, int N, int K, int relu)
{
    const int lane = threadIdx.x & 31;
    const int half = lane >> 4;     // 0: lanes 0-15, 1: lanes 16-31
    const int lm   = lane & 15;
    const int m0 = blockIdx.x << 5;
    const int n0 = blockIdx.y << 5;

    const int akb = half * 8;    // A k-base per ISA 16-bit A 16x32 layout


    const _Float16* xh[2]; const _Float16* xl[2];
    const _Float16* wh[2]; const _Float16* wl[2];
    #pragma unroll
    for (int mt = 0; mt < 2; ++mt) {
        int ar = m0 + mt * 16 + lm;
        if (ar > M - 1) ar = M - 1;            // clamp; garbage rows never stored
        xh[mt] = Xh + (size_t)ar * K;
        xl[mt] = Xl + (size_t)ar * K;
    }
    #pragma unroll
    for (int nt = 0; nt < 2; ++nt) {
        int nr = n0 + nt * 16 + lm;
        wh[nt] = Wh + (size_t)nr * K;
        wl[nt] = Wl + (size_t)nr * K;
    }

    v8f acc00 = {}, acc01 = {}, acc10 = {}, acc11 = {};
    for (int k = 0; k < K; k += 32) {
        v16h ah0 = EPT_cat(*(const v8h*)(xh[0] + k + akb), *(const v8h*)(xh[0] + k + akb + 16));
        v16h al0 = EPT_cat(*(const v8h*)(xl[0] + k + akb), *(const v8h*)(xl[0] + k + akb + 16));
        v16h ah1 = EPT_cat(*(const v8h*)(xh[1] + k + akb), *(const v8h*)(xh[1] + k + akb + 16));
        v16h al1 = EPT_cat(*(const v8h*)(xl[1] + k + akb), *(const v8h*)(xl[1] + k + akb + 16));
        v16h bh0 = EPT_cat(*(const v8h*)(wh[0] + k + akb), *(const v8h*)(wh[0] + k + akb + 16));
        v16h bl0 = EPT_cat(*(const v8h*)(wl[0] + k + akb), *(const v8h*)(wl[0] + k + akb + 16));
        v16h bh1 = EPT_cat(*(const v8h*)(wh[1] + k + akb), *(const v8h*)(wh[1] + k + akb + 16));
        v16h bl1 = EPT_cat(*(const v8h*)(wl[1] + k + akb), *(const v8h*)(wl[1] + k + akb + 16));

        acc00 = __builtin_amdgcn_wmma_f32_16x16x32_f16(false, ah0, false, bh0, (short)0, acc00, false, false);
        acc00 = __builtin_amdgcn_wmma_f32_16x16x32_f16(false, ah0, false, bl0, (short)0, acc00, false, false);
        acc00 = __builtin_amdgcn_wmma_f32_16x16x32_f16(false, al0, false, bh0, (short)0, acc00, false, false);

        acc01 = __builtin_amdgcn_wmma_f32_16x16x32_f16(false, ah0, false, bh1, (short)0, acc01, false, false);
        acc01 = __builtin_amdgcn_wmma_f32_16x16x32_f16(false, ah0, false, bl1, (short)0, acc01, false, false);
        acc01 = __builtin_amdgcn_wmma_f32_16x16x32_f16(false, al0, false, bh1, (short)0, acc01, false, false);

        acc10 = __builtin_amdgcn_wmma_f32_16x16x32_f16(false, ah1, false, bh0, (short)0, acc10, false, false);
        acc10 = __builtin_amdgcn_wmma_f32_16x16x32_f16(false, ah1, false, bl0, (short)0, acc10, false, false);
        acc10 = __builtin_amdgcn_wmma_f32_16x16x32_f16(false, al1, false, bh0, (short)0, acc10, false, false);

        acc11 = __builtin_amdgcn_wmma_f32_16x16x32_f16(false, ah1, false, bh1, (short)0, acc11, false, false);
        acc11 = __builtin_amdgcn_wmma_f32_16x16x32_f16(false, ah1, false, bl1, (short)0, acc11, false, false);
        acc11 = __builtin_amdgcn_wmma_f32_16x16x32_f16(false, al1, false, bh1, (short)0, acc11, false, false);
    }

    // C/D layout: vgpr r -> M = r (+8 for upper lanes); N = lane&15
    #pragma unroll
    for (int nt = 0; nt < 2; ++nt) {
        int n = n0 + nt * 16 + lm;
        float bv = bias[n];
        #pragma unroll
        for (int mt = 0; mt < 2; ++mt) {
            const v8f* accp = mt == 0 ? (nt == 0 ? &acc00 : &acc01)
                                      : (nt == 0 ? &acc10 : &acc11);
            #pragma unroll
            for (int r = 0; r < 8; ++r) {
                int m = m0 + mt * 16 + (half << 3) + r;
                if (m < M) {
                    float v = (*accp)[r] + bv;
                    if (relu) v = fmaxf(v, 0.0f);
                    Y[(size_t)m * N + n] = v;
                }
            }
        }
    }
}

// ---------------------------------------------------------------------------
// out = LayerNorm(a + c) * g + b ; one wave per row of 256
// ---------------------------------------------------------------------------
__global__ void EPT_ln_add(const float* __restrict__ a, const float* __restrict__ c,
                           const float* __restrict__ g, const float* __restrict__ be,
                           float* __restrict__ out, int rows)
{
    int wid = blockIdx.x * (blockDim.x >> 5) + (threadIdx.x >> 5);
    if (wid >= rows) return;
    int lane = threadIdx.x & 31;
    const float* ar = a + (size_t)wid * DM;
    const float* cr = c + (size_t)wid * DM;
    float x[8]; float s = 0.f;
    #pragma unroll
    for (int i = 0; i < 8; ++i) { int j = i*32 + lane; x[i] = ar[j] + cr[j]; s += x[i]; }
    #pragma unroll
    for (int o = 16; o > 0; o >>= 1) s += __shfl_xor(s, o, 32);
    float mean = s * (1.0f/256.0f);
    float v = 0.f;
    #pragma unroll
    for (int i = 0; i < 8; ++i) { float d = x[i]-mean; v += d*d; }
    #pragma unroll
    for (int o = 16; o > 0; o >>= 1) v += __shfl_xor(v, o, 32);
    float inv = rsqrtf(v * (1.0f/256.0f) + 1e-5f);
    float* orow = out + (size_t)wid * DM;
    #pragma unroll
    for (int i = 0; i < 8; ++i) { int j = i*32 + lane; orow[j] = (x[i]-mean)*inv*g[j] + be[j]; }
}

// ---------------------------------------------------------------------------
// [BS,D,H,W] feature maps -> src rows [(b*S+s), D]
// ---------------------------------------------------------------------------
__global__ void EPT_build_src(const float* __restrict__ f0, const float* __restrict__ f1,
                              const float* __restrict__ f2, float* __restrict__ out)
{
    int idx = blockIdx.x * blockDim.x + threadIdx.x;
    if (idx >= BSZ * STOT * DM) return;
    int d = idx & 255; int r = idx >> 8;
    int s = r % STOT; int b = r / STOT;
    const float* f; int HW, sl;
    if (s < 3600)      { f = f0; HW = 3600; sl = s; }
    else if (s < 4500) { f = f1; HW = 900;  sl = s - 3600; }
    else               { f = f2; HW = 225;  sl = s - 4500; }
    out[idx] = f[((size_t)b * DM + d) * HW + sl];
}

// ---------------------------------------------------------------------------
// Sine positional embedding + level embedding -> pos[S, D]
// ---------------------------------------------------------------------------
__global__ void EPT_pos_embed(const float* __restrict__ level_embed, float* __restrict__ pos)
{
    int idx = blockIdx.x * blockDim.x + threadIdx.x;
    if (idx >= STOT * DM) return;
    int d = idx & 255; int s = idx >> 8;
    int Wl, s0, l;
    if (s < 3600)      { Wl = 60; s0 = 0;    l = 0; }
    else if (s < 4500) { Wl = 30; s0 = 3600; l = 1; }
    else               { Wl = 15; s0 = 4500; l = 2; }
    int sl = s - s0; int y = sl / Wl; int x = sl % Wl;
    const float scale = 6.283185307179586f;
    float e; int dd;
    if (d < 128) { e = (float)(y + 1) / ((float)Wl + 1e-6f) * scale; dd = d; }
    else         { e = (float)(x + 1) / ((float)Wl + 1e-6f) * scale; dd = d - 128; }
    int i = dd >> 1;
    float dt = powf(10000.0f, (float)i / 64.0f);
    float arg = e / dt;
    float v = (dd & 1) ? cosf(arg) : sinf(arg);
    pos[idx] = v + level_embed[l * DM + d];
}

// ---------------------------------------------------------------------------
// out[r,d] = x[r,d] + e[erow,d], erow = mode ? r/div : r%div
// ---------------------------------------------------------------------------
__global__ void EPT_bcast_add(const float* __restrict__ x, const float* __restrict__ e,
                              float* __restrict__ out, int rows, int divisor, int mode)
{
    int idx = blockIdx.x * blockDim.x + threadIdx.x;
    if (idx >= rows * DM) return;
    int d = idx & 255; int r = idx >> 8;
    int er = mode ? (r / divisor) : (r % divisor);
    out[idx] = x[idx] + e[(size_t)er * DM + d];
}

// ---------------------------------------------------------------------------
// MSDeformAttn sampling. One wave per (b, q, head); lane = channel (dh=32).
// value: [B, S, NH, 32]; off/aw/out use query rows r = b*strideB + q*strideQ.
// mode 0: encoder refs (per-level grid); mode 1: decoder refs (60x60 grid).
// ---------------------------------------------------------------------------
__global__ void EPT_msda_sample(const float* __restrict__ value, const float* __restrict__ off,
                                const float* __restrict__ aw, float* __restrict__ out,
                                int Lq, int strideB, int strideQ, int mode)
{
    int wid = blockIdx.x * (blockDim.x >> 5) + (threadIdx.x >> 5);
    int lane = threadIdx.x & 31;
    int total = BSZ * Lq * NHEAD;
    if (wid >= total) return;
    int h = wid % NHEAD; int t = wid / NHEAD;
    int q = t % Lq; int b = t / Lq;
    size_t qrow = (size_t)b * strideB + (size_t)q * strideQ;

    float rx, ry;
    if (mode == 0) {
        int Wl, s0;
        if (q < 3600)      { Wl = 60; s0 = 0; }
        else if (q < 4500) { Wl = 30; s0 = 3600; }
        else               { Wl = 15; s0 = 4500; }
        int sl = q - s0;
        rx = ((sl % Wl) + 0.5f) / (float)Wl;
        ry = ((sl / Wl) + 0.5f) / (float)Wl;
    } else {
        rx = ((q % 60) + 0.5f) / 60.0f;
        ry = ((q / 60) + 0.5f) / 60.0f;
    }

    const float* awp = aw + qrow * (NHEAD * NLVL * NPNT) + h * (NLVL * NPNT);
    float wv[12]; float mx = -1e30f;
    #pragma unroll
    for (int j = 0; j < 12; ++j) { wv[j] = awp[j]; mx = fmaxf(mx, wv[j]); }
    float sum = 0.f;
    #pragma unroll
    for (int j = 0; j < 12; ++j) { wv[j] = expf(wv[j] - mx); sum += wv[j]; }
    float invs = 1.0f / sum;

    const float* offp = off + qrow * (NHEAD * NLVL * NPNT * 2) + h * (NLVL * NPNT * 2);
    const int WT[3] = {60, 30, 15};
    const int ST[3] = {0, 3600, 4500};
    float acc = 0.f;
    #pragma unroll
    for (int l = 0; l < NLVL; ++l) {
        int Wl = WT[l], st = ST[l];
        #pragma unroll
        for (int p = 0; p < NPNT; ++p) {
            float ox = offp[(l*NPNT + p)*2 + 0];
            float oy = offp[(l*NPNT + p)*2 + 1];
            // loc = ref + off/[W,H]; pixel = loc*W - 0.5
            float lx = rx * Wl + ox - 0.5f;
            float ly = ry * Wl + oy - 0.5f;
            float wgt = wv[l*NPNT + p] * invs;
            float x0f = floorf(lx), y0f = floorf(ly);
            int x0 = (int)x0f, y0 = (int)y0f;
            float fx = lx - x0f, fy = ly - y0f;
            #pragma unroll
            for (int dy = 0; dy < 2; ++dy) {
                #pragma unroll
                for (int dx = 0; dx < 2; ++dx) {
                    int xi = x0 + dx, yi = y0 + dy;
                    if (xi >= 0 && xi < Wl && yi >= 0 && yi < Wl) {
                        float wc = (dx ? fx : 1.0f - fx) * (dy ? fy : 1.0f - fy);
                        acc += wgt * wc *
                            value[(((size_t)b * STOT + st + yi * Wl + xi) * NHEAD + h) * 32 + lane];
                    }
                }
            }
        }
    }
    out[qrow * DM + h * 32 + lane] = acc;
}

// ---------------------------------------------------------------------------
// Decoder self-attention: L=BS=8 over batch dim, per (q, head, l).
// qkv: [NQ*BS, 768] rows ordered (q*BS + b); out: [NQ*BS, 256].
// ---------------------------------------------------------------------------
__global__ void EPT_mha_attn(const float* __restrict__ qkv, float* __restrict__ out)
{
    int idx = blockIdx.x * blockDim.x + threadIdx.x;
    int total = NQQ * NHEAD * BSZ;
    if (idx >= total) return;
    int l = idx % BSZ; int t = idx / BSZ;
    int h = t % NHEAD; int q = t / NHEAD;

    const float scale = 0.17677669529663687f; // 1/sqrt(32)
    const float* qp = qkv + (size_t)(q * BSZ + l) * 768 + h * 32;
    float qv[32];
    #pragma unroll
    for (int d = 0; d < 32; ++d) qv[d] = qp[d];

    float sc[BSZ];
    float mx = -1e30f;
    for (int s = 0; s < BSZ; ++s) {
        const float* kp = qkv + (size_t)(q * BSZ + s) * 768 + 256 + h * 32;
        float dot = 0.f;
        #pragma unroll
        for (int d = 0; d < 32; ++d) dot += qv[d] * kp[d];
        sc[s] = dot * scale;
        mx = fmaxf(mx, sc[s]);
    }
    float sum = 0.f;
    for (int s = 0; s < BSZ; ++s) { sc[s] = expf(sc[s] - mx); sum += sc[s]; }
    float invs = 1.0f / sum;

    float o[32];
    #pragma unroll
    for (int d = 0; d < 32; ++d) o[d] = 0.f;
    for (int s = 0; s < BSZ; ++s) {
        const float* vp = qkv + (size_t)(q * BSZ + s) * 768 + 512 + h * 32;
        float a = sc[s] * invs;
        #pragma unroll
        for (int d = 0; d < 32; ++d) o[d] += a * vp[d];
    }
    float* op = out + (size_t)(q * BSZ + l) * DM + h * 32;
    #pragma unroll
    for (int d = 0; d < 32; ++d) op[d] = o[d];
}

// ---------------------------------------------------------------------------
extern "C" void kernel_launch(void* const* d_in, const int* in_sizes, int n_in,
                              void* d_out, int out_size, void* d_ws, size_t ws_size,
                              hipStream_t stream)
{
    (void)in_sizes; (void)n_in; (void)out_size; (void)ws_size;

    const float* feat0       = (const float*)d_in[0];
    const float* feat1       = (const float*)d_in[1];
    const float* feat2       = (const float*)d_in[2];
    const float* context     = (const float*)d_in[3];
    const float* query_embed = (const float*)d_in[4];
    const float* level_embed = (const float*)d_in[5];
    const float* enc_off_W = (const float*)d_in[6];   const float* enc_off_b = (const float*)d_in[7];
    const float* enc_aw_W  = (const float*)d_in[8];   const float* enc_aw_b  = (const float*)d_in[9];
    const float* enc_vp_W  = (const float*)d_in[10];  const float* enc_vp_b  = (const float*)d_in[11];
    const float* enc_op_W  = (const float*)d_in[12];  const float* enc_op_b  = (const float*)d_in[13];
    const float* enc_fc1_W = (const float*)d_in[14];  const float* enc_fc1_b = (const float*)d_in[15];
    const float* enc_fc2_W = (const float*)d_in[16];  const float* enc_fc2_b = (const float*)d_in[17];
    const float* enc_ln1_g = (const float*)d_in[18];  const float* enc_ln1_b = (const float*)d_in[19];
    const float* enc_ln2_g = (const float*)d_in[20];  const float* enc_ln2_b = (const float*)d_in[21];
    const float* dec_off_W = (const float*)d_in[22];  const float* dec_off_b = (const float*)d_in[23];
    const float* dec_aw_W  = (const float*)d_in[24];  const float* dec_aw_b  = (const float*)d_in[25];
    const float* dec_vp_W  = (const float*)d_in[26];  const float* dec_vp_b  = (const float*)d_in[27];
    const float* dec_op_W  = (const float*)d_in[28];  const float* dec_op_b  = (const float*)d_in[29];
    const float* dec_fc1_W = (const float*)d_in[30];  const float* dec_fc1_b = (const float*)d_in[31];
    const float* dec_fc2_W = (const float*)d_in[32];  const float* dec_fc2_b = (const float*)d_in[33];
    const float* dec_ln1_g = (const float*)d_in[34];  const float* dec_ln1_b = (const float*)d_in[35];
    const float* dec_ln2_g = (const float*)d_in[36];  const float* dec_ln2_b = (const float*)d_in[37];
    const float* dec_sa_in_W  = (const float*)d_in[38]; const float* dec_sa_in_b  = (const float*)d_in[39];
    const float* dec_sa_out_W = (const float*)d_in[40]; const float* dec_sa_out_b = (const float*)d_in[41];
    const float* dec_ln3_g = (const float*)d_in[42];  const float* dec_ln3_b = (const float*)d_in[43];

    const int MB = BSZ * STOT;  // 37800 encoder rows
    const int MQ = NQQ * BSZ;   // 28800 decoder rows

    float* ws = (float*)d_ws;
    float* bufA   = ws;                        // 9,676,800  enc stream / memory
    float* bufPos = ws + 9676800;              // 1,209,600  pos embedding
    float* bufC   = ws + 10886400;             // 9,676,800  enc query / dec tgt
    float* bufD   = ws + 20563200;             // 9,676,800  value / sa out-proj tmp
    float* bufH   = ws + 30240000;             // 38,707,200 fc1 tmp / qkv / off+aw+samp
    float* bufE   = bufH;                      // off  (<=7,257,600)
    float* bufF   = bufH + 7257600;            // aw   (<=3,628,800)
    float* bufG   = bufH + 10886400;           // samp (<=9,676,800)
    float* bufI   = ws + 68947200;             // 9,676,800  proj/fc2 results
    _Float16* Xhi = (_Float16*)(ws + 78624000);   // 38,707,200 halves
    _Float16* Xlo = (_Float16*)(ws + 97977600);   // 38,707,200 halves
    _Float16* Whi = (_Float16*)(ws + 117331200);  //    262,144 halves
    _Float16* Wlo = (_Float16*)(ws + 117462272);  //    262,144 halves

    auto splitX = [&](const float* X, int n) {
        EPT_split16<<<dim3((n + 255) / 256), dim3(256), 0, stream>>>(X, Xhi, Xlo, n);
    };
    auto splitW = [&](const float* W_, int n) {
        EPT_split16<<<dim3((n + 255) / 256), dim3(256), 0, stream>>>(W_, Whi, Wlo, n);
    };
    auto gemm = [&](const float* W_, const float* b_, float* Y, int M, int N, int K, int relu) {
        splitW(W_, N * K);
        dim3 g((M + 31) / 32, N / 32);
        EPT_gemm_wmma<<<g, dim3(32), 0, stream>>>(Xhi, Xlo, Whi, Wlo, b_, Y, M, N, K, relu);
    };
    auto lnadd = [&](const float* a, const float* c, const float* g_, const float* b_,
                     float* o, int rows) {
        EPT_ln_add<<<dim3((rows + 7) / 8), dim3(256), 0, stream>>>(a, c, g_, b_, o, rows);
    };

    // ---- prep: flatten features, positional embeddings ----
    EPT_build_src<<<dim3((MB * DM + 255) / 256), dim3(256), 0, stream>>>(feat0, feat1, feat2, bufA);
    EPT_pos_embed<<<dim3((STOT * DM + 255) / 256), dim3(256), 0, stream>>>(level_embed, bufPos);

    // ---- encoder ----
    for (int i = 0; i < 3; ++i) {
        const float* offW = enc_off_W + (size_t)i * 192 * 256; const float* offb = enc_off_b + (size_t)i * 192;
        const float* awW  = enc_aw_W  + (size_t)i * 96 * 256;  const float* awb  = enc_aw_b  + (size_t)i * 96;
        const float* vpW  = enc_vp_W  + (size_t)i * 256 * 256; const float* vpb  = enc_vp_b  + (size_t)i * 256;
        const float* opW  = enc_op_W  + (size_t)i * 256 * 256; const float* opb  = enc_op_b  + (size_t)i * 256;
        const float* f1W  = enc_fc1_W + (size_t)i * 1024 * 256; const float* f1b = enc_fc1_b + (size_t)i * 1024;
        const float* f2W  = enc_fc2_W + (size_t)i * 256 * 1024; const float* f2b = enc_fc2_b + (size_t)i * 256;
        const float* l1g = enc_ln1_g + (size_t)i * 256; const float* l1b = enc_ln1_b + (size_t)i * 256;
        const float* l2g = enc_ln2_g + (size_t)i * 256; const float* l2b = enc_ln2_b + (size_t)i * 256;

        EPT_bcast_add<<<dim3((MB * DM + 255) / 256), dim3(256), 0, stream>>>(bufA, bufPos, bufC, MB, STOT, 0);
        splitX(bufA, MB * 256);
        gemm(vpW, vpb, bufD, MB, 256, 256, 0);           // value proj
        splitX(bufC, MB * 256);
        gemm(offW, offb, bufE, MB, 192, 256, 0);         // sampling offsets
        gemm(awW, awb, bufF, MB, 96, 256, 0);            // attention weights
        {
            int waves = BSZ * STOT * NHEAD;
            EPT_msda_sample<<<dim3((waves + 7) / 8), dim3(256), 0, stream>>>(
                bufD, bufE, bufF, bufG, STOT, STOT, 1, 0);
        }
        splitX(bufG, MB * 256);
        gemm(opW, opb, bufI, MB, 256, 256, 0);           // output proj
        lnadd(bufA, bufI, l1g, l1b, bufA, MB);
        splitX(bufA, MB * 256);
        gemm(f1W, f1b, bufH, MB, 1024, 256, 1);          // fc1 + relu
        splitX(bufH, MB * 1024);
        gemm(f2W, f2b, bufI, MB, 256, 1024, 0);          // fc2
        lnadd(bufA, bufI, l2g, l2b, bufA, MB);
    }
    // bufA now holds encoder memory

    // ---- decoder init: tgt = context + query_embed (rows q*BS+b) ----
    EPT_bcast_add<<<dim3((MQ * DM + 255) / 256), dim3(256), 0, stream>>>(context, query_embed, bufC, MQ, BSZ, 1);

    // ---- decoder ----
    for (int i = 0; i < 3; ++i) {
        const float* offW = dec_off_W + (size_t)i * 192 * 256; const float* offb = dec_off_b + (size_t)i * 192;
        const float* awW  = dec_aw_W  + (size_t)i * 96 * 256;  const float* awb  = dec_aw_b  + (size_t)i * 96;
        const float* vpW  = dec_vp_W  + (size_t)i * 256 * 256; const float* vpb  = dec_vp_b  + (size_t)i * 256;
        const float* opW  = dec_op_W  + (size_t)i * 256 * 256; const float* opb  = dec_op_b  + (size_t)i * 256;
        const float* f1W  = dec_fc1_W + (size_t)i * 1024 * 256; const float* f1b = dec_fc1_b + (size_t)i * 1024;
        const float* f2W  = dec_fc2_W + (size_t)i * 256 * 1024; const float* f2b = dec_fc2_b + (size_t)i * 256;
        const float* l1g = dec_ln1_g + (size_t)i * 256; const float* l1b = dec_ln1_b + (size_t)i * 256;
        const float* l2g = dec_ln2_g + (size_t)i * 256; const float* l2b = dec_ln2_b + (size_t)i * 256;
        const float* l3g = dec_ln3_g + (size_t)i * 256; const float* l3b = dec_ln3_b + (size_t)i * 256;
        const float* saW = dec_sa_in_W + (size_t)i * 768 * 256; const float* sab = dec_sa_in_b + (size_t)i * 768;
        const float* soW = dec_sa_out_W + (size_t)i * 256 * 256; const float* sob = dec_sa_out_b + (size_t)i * 256;

        // self-attention (length = BS over batch dim, batch = NQ)
        splitX(bufC, MQ * 256);
        gemm(saW, sab, bufH, MQ, 768, 256, 0);           // fused qkv proj
        EPT_mha_attn<<<dim3((NQQ * NHEAD * BSZ + 255) / 256), dim3(256), 0, stream>>>(bufH, bufI);
        splitX(bufI, MQ * 256);
        gemm(soW, sob, bufD, MQ, 256, 256, 0);           // sa out proj
        lnadd(bufC, bufD, l2g, l2b, bufC, MQ);           // ln2 after self-attn

        // cross deformable attention on encoder memory
        splitX(bufA, MB * 256);
        gemm(vpW, vpb, bufD, MB, 256, 256, 0);           // value proj (memory)
        splitX(bufC, MQ * 256);
        gemm(offW, offb, bufE, MQ, 192, 256, 0);
        gemm(awW, awb, bufF, MQ, 96, 256, 0);
        {
            int waves = BSZ * NQQ * NHEAD;
            EPT_msda_sample<<<dim3((waves + 7) / 8), dim3(256), 0, stream>>>(
                bufD, bufE, bufF, bufG, NQQ, 1, BSZ, 1);
        }
        splitX(bufG, MQ * 256);
        gemm(opW, opb, bufI, MQ, 256, 256, 0);
        lnadd(bufC, bufI, l1g, l1b, bufC, MQ);           // ln1 after msda

        // FFN
        splitX(bufC, MQ * 256);
        gemm(f1W, f1b, bufH, MQ, 1024, 256, 1);
        splitX(bufH, MQ * 1024);
        gemm(f2W, f2b, bufI, MQ, 256, 1024, 0);
        float* dst = (i == 2) ? (float*)d_out : bufC;
        lnadd(bufC, bufI, l3g, l3b, dst, MQ);            // ln3 after FFN
    }
}
